// GQA_38354057953628
// MI455X (gfx1250) — compile-verified, matrix operand order checked by reading, not run on hardware
//
#include <hip/hip_runtime.h>
#include <hip/hip_bf16.h>

#define B_   2
#define T_   2048
#define E_   2048
#define H_   16
#define KVH_ 4
#define D_   128
#define REP_ 4
#define BT_  (B_*T_)
#define KVE_ (KVH_*D_)   // 512

typedef __attribute__((ext_vector_type(8)))  float   v8f;
typedef __attribute__((ext_vector_type(4)))  __bf16  v4bf;
typedef __attribute__((ext_vector_type(8)))  __bf16  v8bf;
typedef __attribute__((ext_vector_type(16))) __bf16  v16bf;

// Load a 16-element bf16 fragment for the 16x16x32 WMMA A/B layout:
// lane l holds row/col (l&15); K elements [k0 + (l>>4)*8, +8) and
// [k0 + 16 + (l>>4)*8, +8). Caller pre-adds row*ld + (l>>4)*8.
static __device__ __forceinline__ v16bf ld_frag(const __bf16* p) {
  v8bf lo = *(const v8bf*)p;
  v8bf hi = *(const v8bf*)(p + 16);
  return __builtin_shufflevector(lo, hi, 0,1,2,3,4,5,6,7,8,9,10,11,12,13,14,15);
}

static __device__ __forceinline__ v8f wmma_bf16(v16bf a, v16bf b, v8f c) {
  return __builtin_amdgcn_wmma_f32_16x16x32_bf16(false, a, false, b, (short)0, c,
                                                 false, false);
}

#define ZERO8 {0.f,0.f,0.f,0.f,0.f,0.f,0.f,0.f}

// One wave computes a 32x64 f32 block: 2 A row-tiles x 4 B col-tiles.
// 12 b128 loads per 8 WMMAs (1.5 loads/WMMA). All fragment loads are issued
// before the WMMA group so the backend can overlap with partial loadcnt waits.
static __device__ __forceinline__ void wave_gemm24(const __bf16* __restrict__ A, int lda,
                                                   const __bf16* __restrict__ Bt, int ldb,
                                                   int K, int row0, int col0, int lane,
                                                   v8f acc[2][4]) {
  const int half = lane >> 4, lr = lane & 15;
  const __bf16* ap0 = A + (size_t)(row0      + lr) * lda + half * 8;
  const __bf16* ap1 = A + (size_t)(row0 + 16 + lr) * lda + half * 8;
  const __bf16* bp0 = Bt + (size_t)(col0      + lr) * ldb + half * 8;
  const __bf16* bp1 = Bt + (size_t)(col0 + 16 + lr) * ldb + half * 8;
  const __bf16* bp2 = Bt + (size_t)(col0 + 32 + lr) * ldb + half * 8;
  const __bf16* bp3 = Bt + (size_t)(col0 + 48 + lr) * ldb + half * 8;
  for (int k0 = 0; k0 < K; k0 += 32) {
    __builtin_prefetch(ap0 + k0 + 512, 0, 3);   // near-cache prefetch, next K-tiles
    __builtin_prefetch(ap1 + k0 + 512, 0, 3);
    v16bf a0 = ld_frag(ap0 + k0);
    v16bf a1 = ld_frag(ap1 + k0);
    v16bf b0 = ld_frag(bp0 + k0);
    v16bf b1 = ld_frag(bp1 + k0);
    v16bf b2 = ld_frag(bp2 + k0);
    v16bf b3 = ld_frag(bp3 + k0);
    acc[0][0] = wmma_bf16(a0, b0, acc[0][0]);
    acc[1][0] = wmma_bf16(a1, b0, acc[1][0]);
    acc[0][1] = wmma_bf16(a0, b1, acc[0][1]);
    acc[1][1] = wmma_bf16(a1, b1, acc[1][1]);
    acc[0][2] = wmma_bf16(a0, b2, acc[0][2]);
    acc[1][2] = wmma_bf16(a1, b2, acc[1][2]);
    acc[0][3] = wmma_bf16(a0, b3, acc[0][3]);
    acc[1][3] = wmma_bf16(a1, b3, acc[1][3]);
  }
}

// ---------------- prep kernels ----------------

__global__ void k_f32_to_bf16(const float* __restrict__ src,
                              __bf16* __restrict__ dst, int n) {
  int i = (blockIdx.x * blockDim.x + threadIdx.x) * 4;
  if (i < n) {
    float4 f = *(const float4*)(src + i);
    v4bf o;
    o[0] = (__bf16)f.x; o[1] = (__bf16)f.y; o[2] = (__bf16)f.z; o[3] = (__bf16)f.w;
    *(v4bf*)(dst + i) = o;
  }
}

// dst[n*K + k] = (bf16) src[k*N + n]   (weights stored KxN row-major)
template <int K, int N>
__global__ void k_transpose_bf16(const float* __restrict__ src,
                                 __bf16* __restrict__ dst) {
  int i = blockIdx.x * blockDim.x + threadIdx.x;
  if (i >= K * N) return;
  int k = i % K;             // K is a power of two -> shifts
  int n = i / K;
  dst[i] = (__bf16)src[(size_t)k * N + n];
}

// ---------------- QKV projections (+ RoPE / transpose) ----------------

#define LOG1E4_OVER_1024 (9.210340371976184f / 1024.0f)

__global__ __launch_bounds__(128)
void k_gemm_rope_q(const __bf16* __restrict__ xb, const __bf16* __restrict__ WT,
                   const float* __restrict__ bias, __bf16* __restrict__ qr) {
  int wid  = (blockIdx.x * blockDim.x + threadIdx.x) >> 5;
  int lane = threadIdx.x & 31;
  const int colGroups = E_ / 64;
  int rt = wid / colGroups, cg = wid % colGroups;
  if (rt >= BT_ / 32) return;
  int row0 = rt * 32, col0 = cg * 64;
  v8f acc[2][4] = {{ZERO8, ZERO8, ZERO8, ZERO8}, {ZERO8, ZERO8, ZERO8, ZERO8}};
  wave_gemm24(xb, E_, WT, E_, E_, row0, col0, lane, acc);
  int half = lane >> 4, lr = lane & 15;
  int b = row0 / T_;
  int tb = row0 % T_;
#pragma unroll
  for (int j = 0; j < 4; ++j) {
    int n = col0 + j * 16 + lr;
    float bv = bias[n];
    int d = n & (D_ - 1);
    int h = n / D_;
    float freq = __expf(-(float)(d >> 1) * LOG1E4_OVER_1024);
#pragma unroll
    for (int r = 0; r < 2; ++r) {
#pragma unroll
      for (int v = 0; v < 8; ++v) {
        int tt = tb + r * 16 + v + 8 * half;
        float ang = (float)tt * freq;
        float c = __cosf(ang), s = __sinf(ang);
        float x = acc[r][j][v] + bv;
        float p = __shfl_xor(x, 1, 32);
        float o = (d & 1) ? (p * s + x * c) : (x * c - p * s);
        qr[(((size_t)b * H_ + h) * T_ + tt) * D_ + d] = (__bf16)o;
      }
    }
  }
}

__global__ __launch_bounds__(128)
void k_gemm_rope_k(const __bf16* __restrict__ xb, const __bf16* __restrict__ WT,
                   const float* __restrict__ bias, __bf16* __restrict__ kr) {
  int wid  = (blockIdx.x * blockDim.x + threadIdx.x) >> 5;
  int lane = threadIdx.x & 31;
  const int colGroups = KVE_ / 64;
  int rt = wid / colGroups, cg = wid % colGroups;
  if (rt >= BT_ / 32) return;
  int row0 = rt * 32, col0 = cg * 64;
  v8f acc[2][4] = {{ZERO8, ZERO8, ZERO8, ZERO8}, {ZERO8, ZERO8, ZERO8, ZERO8}};
  wave_gemm24(xb, E_, WT, E_, E_, row0, col0, lane, acc);
  int half = lane >> 4, lr = lane & 15;
  int b = row0 / T_;
  int tb = row0 % T_;
#pragma unroll
  for (int j = 0; j < 4; ++j) {
    int n = col0 + j * 16 + lr;
    float bv = bias[n];
    int d = n & (D_ - 1);
    int kvh = n / D_;
    float freq = __expf(-(float)(d >> 1) * LOG1E4_OVER_1024);
#pragma unroll
    for (int r = 0; r < 2; ++r) {
#pragma unroll
      for (int v = 0; v < 8; ++v) {
        int tt = tb + r * 16 + v + 8 * half;
        float ang = (float)tt * freq;
        float c = __cosf(ang), s = __sinf(ang);
        float x = acc[r][j][v] + bv;
        float p = __shfl_xor(x, 1, 32);
        float o = (d & 1) ? (p * s + x * c) : (x * c - p * s);
        kr[(((size_t)b * KVH_ + kvh) * T_ + tt) * D_ + d] = (__bf16)o;
      }
    }
  }
}

// V projection, stored transposed: vt[b][kvh][d][t]
__global__ __launch_bounds__(128)
void k_gemm_v(const __bf16* __restrict__ xb, const __bf16* __restrict__ WT,
              const float* __restrict__ bias, __bf16* __restrict__ vt) {
  int wid  = (blockIdx.x * blockDim.x + threadIdx.x) >> 5;
  int lane = threadIdx.x & 31;
  const int colGroups = KVE_ / 64;
  int rt = wid / colGroups, cg = wid % colGroups;
  if (rt >= BT_ / 32) return;
  int row0 = rt * 32, col0 = cg * 64;
  v8f acc[2][4] = {{ZERO8, ZERO8, ZERO8, ZERO8}, {ZERO8, ZERO8, ZERO8, ZERO8}};
  wave_gemm24(xb, E_, WT, E_, E_, row0, col0, lane, acc);
  int half = lane >> 4, lr = lane & 15;
  int b = row0 / T_;
  int tb = row0 % T_;
#pragma unroll
  for (int j = 0; j < 4; ++j) {
    int n = col0 + j * 16 + lr;
    float bv = bias[n];
    int d = n & (D_ - 1);
    int kvh = n >> 7;
#pragma unroll
    for (int r = 0; r < 2; ++r) {
      v8bf pk;
#pragma unroll
      for (int v = 0; v < 8; ++v) pk[v] = (__bf16)(acc[r][j][v] + bv);
      size_t idx = (((size_t)b * KVH_ + kvh) * D_ + d) * T_ + tb + r * 16 + 8 * half;
      *(v8bf*)(vt + idx) = pk;   // contiguous 8 t-values per lane
    }
  }
}

// ---------------- flash attention (one wave = 16 query rows) ----------------

__global__ __launch_bounds__(128)
void k_attn(const __bf16* __restrict__ qr, const __bf16* __restrict__ kr,
            const __bf16* __restrict__ vt, __bf16* __restrict__ ob) {
  __shared__ __bf16 plds[4][16][32];
  int wv   = threadIdx.x >> 5;
  int lane = threadIdx.x & 31;
  int w = blockIdx.x * 4 + wv;
  const int T16 = T_ / 16;
  int ttile = w % T16;
  int h = (w / T16) % H_;
  int b = w / (T16 * H_);
  int t0 = ttile * 16;
  int kvh = h / REP_;
  int half = lane >> 4, lr = lane & 15;

  const __bf16* qp    = qr + (((size_t)b * H_ + h) * T_ + t0 + lr) * D_ + half * 8;
  const __bf16* kbase = kr + ((size_t)b * KVH_ + kvh) * T_ * D_;
  const __bf16* vbase = vt + ((size_t)b * KVH_ + kvh) * (size_t)D_ * T_;

  v16bf qf[4];
#pragma unroll
  for (int i = 0; i < 4; ++i) qf[i] = ld_frag(qp + i * 32);

  v8f acc[8];
#pragma unroll
  for (int j = 0; j < 8; ++j) acc[j] = (v8f)ZERO8;
  float mrow[8], lsum[8];
#pragma unroll
  for (int v = 0; v < 8; ++v) { mrow[v] = -1e30f; lsum[v] = 0.f; }

  for (int s0 = 0; s0 < t0 + 16; s0 += 32) {
    v8f sa0 = ZERO8;
    v8f sa1 = ZERO8;
    const __bf16* kp0 = kbase + (size_t)(s0 + lr) * D_ + half * 8;
    const __bf16* kp1 = kbase + (size_t)(s0 + 16 + lr) * D_ + half * 8;
#pragma unroll
    for (int i = 0; i < 4; ++i) {
      v16bf kf0 = ld_frag(kp0 + i * 32);
      v16bf kf1 = ld_frag(kp1 + i * 32);
      sa0 = wmma_bf16(qf[i], kf0, sa0);
      sa1 = wmma_bf16(qf[i], kf1, sa1);
    }
    if (s0 + 31 > t0) {          // causal mask needed only near diagonal
#pragma unroll
      for (int v = 0; v < 8; ++v) {
        int t = t0 + v + 8 * half;
        if (s0 + lr      > t) sa0[v] = -1e30f;
        if (s0 + 16 + lr > t) sa1[v] = -1e30f;
      }
    }
    // online softmax: row stats live in VGPR index v, cols across 16 lanes
#pragma unroll
    for (int v = 0; v < 8; ++v) {
      float loc = fmaxf(sa0[v], sa1[v]);
      loc = fmaxf(loc, __shfl_xor(loc, 1, 32));
      loc = fmaxf(loc, __shfl_xor(loc, 2, 32));
      loc = fmaxf(loc, __shfl_xor(loc, 4, 32));
      loc = fmaxf(loc, __shfl_xor(loc, 8, 32));
      float mnew = fmaxf(mrow[v], loc);
      float sc = __expf(mrow[v] - mnew);
      mrow[v] = mnew;
      lsum[v] *= sc;
#pragma unroll
      for (int j = 0; j < 8; ++j) acc[j][v] *= sc;
      float p0 = __expf(sa0[v] - mnew);
      float p1 = __expf(sa1[v] - mnew);
      float rs = p0 + p1;
      rs += __shfl_xor(rs, 1, 32);
      rs += __shfl_xor(rs, 2, 32);
      rs += __shfl_xor(rs, 4, 32);
      rs += __shfl_xor(rs, 8, 32);
      lsum[v] += rs;
      int m = v + 8 * half;
      plds[wv][m][lr]      = (__bf16)p0;   // D-layout -> (m, s_local) in LDS
      plds[wv][m][16 + lr] = (__bf16)p1;
    }
    asm volatile("s_wait_dscnt 0x0" ::: "memory");  // per-wave LDS RAW fence
    v16bf pf = ld_frag(&plds[wv][lr][half * 8]);    // A-layout reload
#pragma unroll
    for (int j = 0; j < 8; j += 2) {
      v16bf bv0 = ld_frag(vbase + (size_t)((j    ) * 16 + lr) * T_ + s0 + half * 8);
      v16bf bv1 = ld_frag(vbase + (size_t)((j + 1) * 16 + lr) * T_ + s0 + half * 8);
      acc[j]     = wmma_bf16(pf, bv0, acc[j]);
      acc[j + 1] = wmma_bf16(pf, bv1, acc[j + 1]);
    }
  }

  const float post_scale = 0.022097086912079608f;  // E^-0.5, applied after softmax
#pragma unroll
  for (int j = 0; j < 8; ++j) {
    int dcol = j * 16 + lr;
#pragma unroll
    for (int v = 0; v < 8; ++v) {
      int t = t0 + v + 8 * half;
      float o = acc[j][v] / lsum[v] * post_scale;
      ob[((size_t)(b * T_ + t)) * E_ + h * D_ + dcol] = (__bf16)o;
    }
  }
}

// ---------------- output projection ----------------

__global__ __launch_bounds__(128)
void k_gemm_out(const __bf16* __restrict__ ob, const __bf16* __restrict__ WoT,
                const float* __restrict__ bias, float* __restrict__ out) {
  int wid  = (blockIdx.x * blockDim.x + threadIdx.x) >> 5;
  int lane = threadIdx.x & 31;
  const int colGroups = E_ / 64;
  int rt = wid / colGroups, cg = wid % colGroups;
  if (rt >= BT_ / 32) return;
  int row0 = rt * 32, col0 = cg * 64;
  v8f acc[2][4] = {{ZERO8, ZERO8, ZERO8, ZERO8}, {ZERO8, ZERO8, ZERO8, ZERO8}};
  wave_gemm24(ob, E_, WoT, E_, E_, row0, col0, lane, acc);
  int half = lane >> 4, lr = lane & 15;
#pragma unroll
  for (int j = 0; j < 4; ++j) {
    int n = col0 + j * 16 + lr;
    float bv = bias[n];
#pragma unroll
    for (int r = 0; r < 2; ++r) {
#pragma unroll
      for (int v = 0; v < 8; ++v)
        out[(size_t)(row0 + r * 16 + v + 8 * half) * E_ + n] = acc[r][j][v] + bv;
    }
  }
}

// ---------------- host launcher ----------------

extern "C" void kernel_launch(void* const* d_in, const int* in_sizes, int n_in,
                              void* d_out, int out_size, void* d_ws, size_t ws_size,
                              hipStream_t stream) {
  const float* x  = (const float*)d_in[0];
  const float* Wq = (const float*)d_in[1];
  const float* bq = (const float*)d_in[2];
  const float* Wk = (const float*)d_in[3];
  const float* bk = (const float*)d_in[4];
  const float* Wv = (const float*)d_in[5];
  const float* bv = (const float*)d_in[6];
  const float* Wo = (const float*)d_in[7];
  const float* bo = (const float*)d_in[8];
  float* out = (float*)d_out;

  char* ws = (char*)d_ws;
  __bf16* xb  = (__bf16*)(ws + 0);          // 16 MB  (BT x E)
  __bf16* WqT = (__bf16*)(ws + 16777216);   //  8 MB  (E x E, [n][k])
  __bf16* WkT = (__bf16*)(ws + 25165824);   //  2 MB  (512 x E)
  __bf16* WvT = (__bf16*)(ws + 27262976);   //  2 MB
  __bf16* WoT = (__bf16*)(ws + 29360128);   //  8 MB
  __bf16* qr  = (__bf16*)(ws + 37748736);   // 16 MB  [B][H][T][D]
  __bf16* kr  = (__bf16*)(ws + 54525952);   //  4 MB  [B][KVH][T][D]
  __bf16* vt  = (__bf16*)(ws + 58720256);   //  4 MB  [B][KVH][D][T]
  __bf16* ob  = (__bf16*)(ws + 62914560);   // 16 MB  [B][T][E]
  (void)ws_size; (void)in_sizes; (void)n_in; (void)out_size;

  { int n = BT_ * E_;
    k_f32_to_bf16<<<(n / 4 + 255) / 256, 256, 0, stream>>>(x, xb, n); }

  k_transpose_bf16<E_, E_>  <<<(E_ * E_   + 255) / 256, 256, 0, stream>>>(Wq, WqT);
  k_transpose_bf16<E_, KVE_><<<(E_ * KVE_ + 255) / 256, 256, 0, stream>>>(Wk, WkT);
  k_transpose_bf16<E_, KVE_><<<(E_ * KVE_ + 255) / 256, 256, 0, stream>>>(Wv, WvT);
  k_transpose_bf16<E_, E_>  <<<(E_ * E_   + 255) / 256, 256, 0, stream>>>(Wo, WoT);

  k_gemm_rope_q<<<(BT_ / 32) * (E_ / 64)   / 4, 128, 0, stream>>>(xb, WqT, bq, qr);
  k_gemm_rope_k<<<(BT_ / 32) * (KVE_ / 64) / 4, 128, 0, stream>>>(xb, WkT, bk, kr);
  k_gemm_v     <<<(BT_ / 32) * (KVE_ / 64) / 4, 128, 0, stream>>>(xb, WvT, bv, vt);

  k_attn<<<(B_ * H_ * (T_ / 16)) / 4, 128, 0, stream>>>(qr, kr, vt, ob);

  k_gemm_out<<<(BT_ / 32) * (E_ / 64) / 4, 128, 0, stream>>>(ob, WoT, bo, out);
}
